// VoxelDecoder_77343771066949
// MI455X (gfx1250) — matrix operand order repaired and checked by reading, not compiled, not run on hardware
//
#include <hip/hip_runtime.h>
#include <hip/hip_bf16.h>

typedef __attribute__((ext_vector_type(16))) __bf16 v16bf;
typedef __attribute__((ext_vector_type(8)))  float  v8f;
typedef __attribute__((ext_vector_type(8)))  unsigned int v8u;

// fp32 -> bf16 bits, round-to-nearest-even
__device__ __forceinline__ unsigned short f2bf_bits(float f) {
  unsigned u = __builtin_bit_cast(unsigned, f);
  unsigned r = u + 0x7FFFu + ((u >> 16) & 1u);
  return (unsigned short)(r >> 16);
}
__device__ __forceinline__ unsigned pack2(float lo, float hi) {
  return (unsigned)f2bf_bits(lo) | ((unsigned)f2bf_bits(hi) << 16);
}

__global__ __launch_bounds__(256) void fill_zero(float* p, int n) {
  int i = blockIdx.x * 256 + threadIdx.x;
  if (i < n) p[i] = 0.0f;
}

// Repack fp32 weights [K, Cin, Cout] -> bf16 pair-major uints: Wp[k][kp][col] = {W[k][2kp][col], W[k][2kp+1][col]}
__global__ __launch_bounds__(256) void pack_weights(const float* __restrict__ W,
                                                    int K, int Cin, int Cout,
                                                    unsigned* __restrict__ Wp) {
  int Cinp = Cin >> 1;
  long total = (long)K * Cinp * Cout;
  long i = (long)blockIdx.x * 256 + threadIdx.x;
  if (i < total) {
    int col = (int)(i % Cout);
    long t  = i / Cout;
    int kp  = (int)(t % Cinp);
    int k   = (int)(t / Cinp);
    long base = ((long)k * Cin + 2 * kp) * Cout + col;
    Wp[i] = pack2(W[base], W[base + Cout]);
  }
}

// fp32 [N, Cs] -> bf16-pair uints written at dst[r*lddp + p] (dst pre-offset in pairs)
__global__ __launch_bounds__(256) void copy_cols_pair(const float* __restrict__ src, int N, int Cs,
                                                      unsigned* __restrict__ dst, int lddp) {
  int Csp = Cs >> 1;
  int i = blockIdx.x * 256 + threadIdx.x;
  if (i < N * Csp) {
    int r = i / Csp, p = i - r * Csp;
    dst[(long)r * lddp + p] = pack2(src[(long)r * Cs + 2 * p], src[(long)r * Cs + 2 * p + 1]);
  }
}

// Gather -> bf16 WMMA GEMM -> scatter-add.
// Block: 256 threads = 8 waves; tile = 32 rows x 256 cols; each wave: 32x32 (4 WMMAs).
// Xp: bf16-pair activations [*, Cinp]; Wp: pair-major packed weights [K, Cinp, Cout].
// in_idx/out_idx == nullptr => dense identity (plain store).
__global__ __launch_bounds__(256) void spconv_wmma(
    const unsigned* __restrict__ Xp,
    const unsigned* __restrict__ Wp,
    const int* __restrict__ in_idx,   // [K, M] or null
    const int* __restrict__ out_idx,  // [K, M] or null
    int M, int Cinp, int Cout,
    float* __restrict__ out)          // [Nout, Cout] fp32 (pre-zeroed when scattering)
{
  __shared__ unsigned As[32][18];     // 32 rows x 16 K-pairs (K-chunk = 32); stride 18 = conflict-free
  __shared__ unsigned Bs[16][260];    // 16 K-pairs x 256 cols (padded stride)

  const int tid  = threadIdx.x;
  const int lane = tid & 31;
  const int wave = tid >> 5;
  const int grp  = lane >> 4;
  const int lnm  = lane & 15;

  const int m0     = blockIdx.x * 32;
  const int n0base = blockIdx.y * 256;
  const int kk     = blockIdx.z;

  const long wbase = (long)kk * Cinp * Cout;
  const int* inI   = in_idx  ? (in_idx  + kk * M) : nullptr;
  const int* outI  = out_idx ? (out_idx + kk * M) : nullptr;

  // A staging: each thread owns rows (ar) and (ar+16), one pair column each
  const int ar = tid >> 4;
  const int ap = tid & 15;
  const int mr0 = m0 + ar;
  const int mr1 = m0 + ar + 16;
  const bool vr0 = mr0 < M;
  const bool vr1 = mr1 < M;
  const long ab0 = (long)(vr0 ? (inI ? inI[mr0] : mr0) : 0) * Cinp;
  const long ab1 = (long)(vr1 ? (inI ? inI[mr1] : mr1) : 0) * Cinp;

  v8f acc00 = {}, acc01 = {}, acc10 = {}, acc11 = {};

  const int nchunks = (Cinp + 15) >> 4;
  for (int ch = 0; ch < nchunks; ++ch) {
    const int p0 = ch << 4;   // first K-pair of this chunk
    // --- stage A: 32 (gathered) rows x 16 pairs; OOB rows zeroed ---
    {
      int pidx = p0 + ap;
      bool pv = pidx < Cinp;
      As[ar][ap]      = (pv && vr0) ? Xp[ab0 + pidx] : 0u;
      As[ar + 16][ap] = (pv && vr1) ? Xp[ab1 + pidx] : 0u;
    }
    // --- stage B: 16 K-pairs x 256 cols (16 uints per thread, coalesced) ---
#pragma unroll
    for (int s = 0; s < 16; ++s) {
      int i  = tid + s * 256;
      int kp = i >> 8;
      int c  = i & 255;
      int gkp = p0 + kp;
      int col = n0base + c;
      Bs[kp][c] = (gkp < Cinp && col < Cout) ? Wp[wbase + (long)gkp * Cout + col] : 0u;
    }
    __syncthreads();

    // --- fragments per CDNA5 16-bit A(16x32)/B(32x16) VGPR layouts ---
    v8u au0, au1, bu0, bu1;
#pragma unroll
    for (int j = 0; j < 8; ++j) {
      int pidx = grp * 4 + j + ((j >= 4) ? 4 : 0);
      au0[j] = As[lnm][pidx];
      au1[j] = As[lnm + 16][pidx];
      bu0[j] = Bs[pidx][wave * 32 + lnm];
      bu1[j] = Bs[pidx][wave * 32 + 16 + lnm];
    }
    v16bf a0 = __builtin_bit_cast(v16bf, au0);
    v16bf a1 = __builtin_bit_cast(v16bf, au1);
    v16bf b0 = __builtin_bit_cast(v16bf, bu0);
    v16bf b1 = __builtin_bit_cast(v16bf, bu1);
    acc00 = __builtin_amdgcn_wmma_f32_16x16x32_bf16(false, a0, false, b0, (short)0, acc00, false, false);
    acc01 = __builtin_amdgcn_wmma_f32_16x16x32_bf16(false, a0, false, b1, (short)0, acc01, false, false);
    acc10 = __builtin_amdgcn_wmma_f32_16x16x32_bf16(false, a1, false, b0, (short)0, acc10, false, false);
    acc11 = __builtin_amdgcn_wmma_f32_16x16x32_bf16(false, a1, false, b1, (short)0, acc11, false, false);
    __syncthreads();
  }

  // --- C layout: lanes 0-15 vgpr v -> (M=v, N=lane); lanes 16-31 -> (M=8+v, N=lane-16)
  const int nc0 = n0base + wave * 32 + lnm;
  const int nc1 = nc0 + 16;
  const bool cv0 = nc0 < Cout;
  const bool cv1 = nc1 < Cout;
  if (outI) {
#pragma unroll
    for (int v = 0; v < 8; ++v) {
      int m = m0 + v + grp * 8;
      if (m < M) {
        long rb = (long)outI[m] * Cout;
        if (cv0) unsafeAtomicAdd(&out[rb + nc0], acc00[v]);
        if (cv1) unsafeAtomicAdd(&out[rb + nc1], acc01[v]);
      }
      int m2 = m + 16;
      if (m2 < M) {
        long rb = (long)outI[m2] * Cout;
        if (cv0) unsafeAtomicAdd(&out[rb + nc0], acc10[v]);
        if (cv1) unsafeAtomicAdd(&out[rb + nc1], acc11[v]);
      }
    }
  } else {
#pragma unroll
    for (int v = 0; v < 8; ++v) {
      int m = m0 + v + grp * 8;
      if (m < M) {
        long rb = (long)m * Cout;
        if (cv0) out[rb + nc0] = acc00[v];
        if (cv1) out[rb + nc1] = acc01[v];
      }
      int m2 = m + 16;
      if (m2 < M) {
        long rb = (long)m2 * Cout;
        if (cv0) out[rb + nc0] = acc10[v];
        if (cv1) out[rb + nc1] = acc11[v];
      }
    }
  }
}

// coalesced per-row-block partial sums: threads map to consecutive channels
__global__ __launch_bounds__(256) void bn_partial(const float* __restrict__ x, int N, int C,
                                                  int rows_per_block, float* __restrict__ accum) {
  int r0 = blockIdx.x * rows_per_block;
  int r1 = r0 + rows_per_block; if (r1 > N) r1 = N;
  for (int c = threadIdx.x; c < C; c += 256) {
    float s = 0.0f, q = 0.0f;
    for (int r = r0; r < r1; ++r) {
      float v = x[(long)r * C + c];
      s += v; q += v * v;
    }
    unsafeAtomicAdd(&accum[c], s);
    unsafeAtomicAdd(&accum[C + c], q);
  }
}

__global__ __launch_bounds__(256) void bn_finalize(const float* __restrict__ accum, int N, int C,
                                                   float* __restrict__ stats) {
  int c = blockIdx.x * 256 + threadIdx.x;
  if (c < C) {
    float mean = accum[c] / (float)N;
    float var  = accum[C + c] / (float)N - mean * mean;
    stats[c]     = mean;
    stats[C + c] = rsqrtf(var + 1e-5f);
  }
}

// normalize + ReLU -> bf16-pair activations (feeds next conv's gather)
__global__ __launch_bounds__(256) void bn_apply_pair(const float* __restrict__ x, int N, int C,
                                                     const float* __restrict__ g,
                                                     const float* __restrict__ bta,
                                                     const float* __restrict__ stats,
                                                     unsigned* __restrict__ dst, int lddp) {
  int Cp = C >> 1;
  int i = blockIdx.x * 256 + threadIdx.x;
  if (i < N * Cp) {
    int r = i / Cp, p = i - r * Cp;
    int c0 = 2 * p, c1 = c0 + 1;
    long rb = (long)r * C;
    float y0 = fmaxf((x[rb + c0] - stats[c0]) * stats[C + c0] * g[c0] + bta[c0], 0.0f);
    float y1 = fmaxf((x[rb + c1] - stats[c1]) * stats[C + c1] * g[c1] + bta[c1], 0.0f);
    dst[(long)r * lddp + p] = pack2(y0, y1);
  }
}

// normalize + ReLU -> fp32 (final output)
__global__ __launch_bounds__(256) void bn_apply_f32(const float* __restrict__ x, int N, int C,
                                                    const float* __restrict__ g,
                                                    const float* __restrict__ bta,
                                                    const float* __restrict__ stats,
                                                    float* __restrict__ dst) {
  int i = blockIdx.x * 256 + threadIdx.x;
  if (i < N * C) {
    int c = i % C;
    float y = (x[i] - stats[c]) * stats[C + c] * g[c] + bta[c];
    dst[i] = fmaxf(y, 0.0f);
  }
}

// ---------------- host orchestration ----------------

static void conv_layer(const unsigned* Xp, const unsigned* Wp, const int* ii, const int* oi,
                       int M, int K, int Cinp, int Cout, int Nout, float* acc, hipStream_t s) {
  if (oi) {
    int n = Nout * Cout;
    fill_zero<<<dim3((n + 255) / 256), dim3(256), 0, s>>>(acc, n);
  }
  dim3 grid((M + 31) / 32, (Cout + 255) / 256, K);
  spconv_wmma<<<grid, dim3(256), 0, s>>>(Xp, Wp, ii, oi, M, Cinp, Cout, acc);
}

static void bn_stats(const float* x, int N, int C, float* accum, float* stats, hipStream_t s) {
  fill_zero<<<dim3((2 * C + 255) / 256), dim3(256), 0, s>>>(accum, 2 * C);
  const int RB = 128;
  bn_partial<<<dim3((N + RB - 1) / RB), dim3(256), 0, s>>>(x, N, C, RB, accum);
  bn_finalize<<<dim3((C + 255) / 256), dim3(256), 0, s>>>(accum, N, C, stats);
}

static void pack_w(const float* W, int K, int Cin, int Cout, unsigned* Wp, hipStream_t s) {
  long total = (long)K * (Cin >> 1) * Cout;
  pack_weights<<<dim3((unsigned)((total + 255) / 256)), dim3(256), 0, s>>>(W, K, Cin, Cout, Wp);
}

extern "C" void kernel_launch(void* const* d_in, const int* in_sizes, int n_in,
                              void* d_out, int out_size, void* d_ws, size_t ws_size,
                              hipStream_t stream) {
  const int N0 = 30000, N1 = 12000, N2 = 5000, N3 = 2000, K = 27;
  const int M0 = 10000, M1 = 4000, M2 = 1600;

  const float* f0 = (const float*)d_in[0];
  const float* f1 = (const float*)d_in[1];
  const float* f2 = (const float*)d_in[2];
  const float* f3 = (const float*)d_in[3];
  const float* W_up2 = (const float*)d_in[4];
  const float* W_up1 = (const float*)d_in[5];
  const float* W_up0 = (const float*)d_in[6];
  const float* W_s0  = (const float*)d_in[7];
  const float* W_s1  = (const float*)d_in[8];
  const float* W_s2  = (const float*)d_in[9];
  const float* g_up2 = (const float*)d_in[10]; const float* b_up2 = (const float*)d_in[11];
  const float* g_up1 = (const float*)d_in[12]; const float* b_up1 = (const float*)d_in[13];
  const float* g_up0 = (const float*)d_in[14]; const float* b_up0 = (const float*)d_in[15];
  const float* g_s0  = (const float*)d_in[16]; const float* b_s0  = (const float*)d_in[17];
  const float* g_s1  = (const float*)d_in[18]; const float* b_s1  = (const float*)d_in[19];
  const float* g_s2  = (const float*)d_in[20]; const float* b_s2  = (const float*)d_in[21];
  const int* up2_in = (const int*)d_in[22]; const int* up2_out = (const int*)d_in[23];
  const int* up1_in = (const int*)d_in[24]; const int* up1_out = (const int*)d_in[25];
  const int* up0_in = (const int*)d_in[26]; const int* up0_out = (const int*)d_in[27];
  const int* sm0_in = (const int*)d_in[28]; const int* sm0_out = (const int*)d_in[29];
  const int* sm1_in = (const int*)d_in[30]; const int* sm1_out = (const int*)d_in[31];

  // workspace layout
  float* ws = (float*)d_ws;
  size_t o = 0;
  float* ACC   = ws + o; o += (size_t)N0 * 512;   // shared pre-BN fp32 accumulator
  float* ACCUM = ws + o; o += 1024;               // bn partial sums
  float* ST    = ws + o; o += 1024;               // bn stats (mean | inv-std)
  unsigned* wsu = (unsigned*)(ws + o);
  size_t u = 0;
  unsigned* Wp_up2 = wsu + u; u += (size_t)K * 64  * 128;
  unsigned* Wp_up1 = wsu + u; u += (size_t)K * 96  * 192;
  unsigned* Wp_up0 = wsu + u; u += (size_t)K * 112 * 224;
  unsigned* Wp_s0  = wsu + u; u += (size_t)K * 120 * 512;
  unsigned* Wp_s1  = wsu + u; u += (size_t)K * 256 * 512;
  unsigned* Wp_s2  = wsu + u; u += (size_t)256 * 512;
  unsigned* F3b    = wsu + u; u += (size_t)N3 * 64;
  unsigned* X2b    = wsu + u; u += (size_t)N2 * 96;
  unsigned* X1b    = wsu + u; u += (size_t)N1 * 112;
  unsigned* X0b    = wsu + u; u += (size_t)N0 * 120;
  unsigned* S0b    = wsu + u; u += (size_t)N0 * 256;
  unsigned* S1b    = wsu + u; u += (size_t)N0 * 256;

  // one-time (per launch) bf16 pair-major packing of weights + f3
  pack_w(W_up2, K, 128, 128, Wp_up2, stream);
  pack_w(W_up1, K, 192, 192, Wp_up1, stream);
  pack_w(W_up0, K, 224, 224, Wp_up0, stream);
  pack_w(W_s0,  K, 240, 512, Wp_s0,  stream);
  pack_w(W_s1,  K, 512, 512, Wp_s1,  stream);
  pack_w(W_s2,  1, 512, 512, Wp_s2,  stream);
  copy_cols_pair<<<dim3((N3 * 64 + 255) / 256), dim3(256), 0, stream>>>(f3, N3, 128, F3b, 64);

  // --- decoder level 2: f3 -> N2, Cout=128, concat f2 (64) ---
  conv_layer(F3b, Wp_up2, up2_in, up2_out, M2, K, 64, 128, N2, ACC, stream);
  bn_stats(ACC, N2, 128, ACCUM, ST, stream);
  bn_apply_pair<<<dim3((N2 * 64 + 255) / 256), dim3(256), 0, stream>>>(ACC, N2, 128, g_up2, b_up2, ST, X2b, 96);
  copy_cols_pair<<<dim3((N2 * 32 + 255) / 256), dim3(256), 0, stream>>>(f2, N2, 64, X2b + 64, 96);

  // --- decoder level 1: X2 -> N1, Cout=192, concat f1 (32) ---
  conv_layer(X2b, Wp_up1, up1_in, up1_out, M1, K, 96, 192, N1, ACC, stream);
  bn_stats(ACC, N1, 192, ACCUM, ST, stream);
  bn_apply_pair<<<dim3((N1 * 96 + 255) / 256), dim3(256), 0, stream>>>(ACC, N1, 192, g_up1, b_up1, ST, X1b, 112);
  copy_cols_pair<<<dim3((N1 * 16 + 255) / 256), dim3(256), 0, stream>>>(f1, N1, 32, X1b + 96, 112);

  // --- decoder level 0: X1 -> N0, Cout=224, concat f0 (16) ---
  conv_layer(X1b, Wp_up0, up0_in, up0_out, M0, K, 112, 224, N0, ACC, stream);
  bn_stats(ACC, N0, 224, ACCUM, ST, stream);
  bn_apply_pair<<<dim3((N0 * 112 + 255) / 256), dim3(256), 0, stream>>>(ACC, N0, 224, g_up0, b_up0, ST, X0b, 120);
  copy_cols_pair<<<dim3((N0 * 8 + 255) / 256), dim3(256), 0, stream>>>(f0, N0, 16, X0b + 112, 120);

  // --- smooth 0: X0 (240) -> 512 ---
  conv_layer(X0b, Wp_s0, sm0_in, sm0_out, M0, K, 120, 512, N0, ACC, stream);
  bn_stats(ACC, N0, 512, ACCUM, ST, stream);
  bn_apply_pair<<<dim3((N0 * 256 + 255) / 256), dim3(256), 0, stream>>>(ACC, N0, 512, g_s0, b_s0, ST, S0b, 256);

  // --- smooth 1: S0 (512) -> 512 ---
  conv_layer(S0b, Wp_s1, sm1_in, sm1_out, M0, K, 256, 512, N0, ACC, stream);
  bn_stats(ACC, N0, 512, ACCUM, ST, stream);
  bn_apply_pair<<<dim3((N0 * 256 + 255) / 256), dim3(256), 0, stream>>>(ACC, N0, 512, g_s1, b_s1, ST, S1b, 256);

  // --- pointwise: S1 @ W_s2 (dense, K=1) ---
  conv_layer(S1b, Wp_s2, nullptr, nullptr, N0, 1, 256, 512, N0, ACC, stream);
  bn_stats(ACC, N0, 512, ACCUM, ST, stream);
  bn_apply_f32<<<dim3((N0 * 512 + 255) / 256), dim3(256), 0, stream>>>(ACC, N0, 512, g_s2, b_s2, ST, (float*)d_out);
}
